// Embedder_17592186044591
// MI455X (gfx1250) — matrix operand order repaired and checked by reading, not compiled, not run on hardware
//
#include <hip/hip_runtime.h>
#include <hip/hip_bf16.h>

// ---------------------------------------------------------------------------
// MI455X (gfx1250) multi-head attention block, wave32 + v_wmma_f32_16x16x32_bf16
// + async global->LDS staging (ASYNCcnt) + global_prefetch.
//
//   q,k,v = x@W{q,k,v} + b{q,k,v}        [S,E] -> [H,S,d], H=16, d=64
//   scores = q.kT (NO 1/sqrt(d));  out-of-segment scores set to 0.0
//   weights = softmax(scores) over ALL keys (zeros participate)
//   out = weights.v -> [S,E]
//   result = out@Wo + bo ; return mean over rows with seg==seg[pos]
//
// The readout is linear: mean(out@Wo+bo) = mean(out)@Wo+bo, so the final GEMM
// collapses to a single E-vector GEMV.
// ---------------------------------------------------------------------------

#define S_DIM 2048
#define E_DIM 1024
#define H_DIM 16
#define D_DIM 64

typedef __bf16 bf16_t;
typedef bf16_t v16bf __attribute__((ext_vector_type(16)));
typedef float  v8f   __attribute__((ext_vector_type(8)));
typedef unsigned int v8ui __attribute__((ext_vector_type(8)));

struct FragBits { uint4 lo, hi; };  // 32B, bit-identical to v16bf

// Build a 16-element bf16 fragment from two contiguous 16B chunks.
// ext_vector is little-endian: element e lives in half-dword e, so the
// concatenation {lo,hi} maps exactly to elements 0..7 / 8..15.
static __device__ __forceinline__ v16bf make_frag(uint4 lo, uint4 hi) {
    FragBits b{lo, hi};
    return __builtin_bit_cast(v16bf, b);
}
static __device__ __forceinline__ unsigned short f32_to_bf16_rne(float f) {
    unsigned u = __builtin_bit_cast(unsigned, f);
    unsigned r = u + 0x7FFFu + ((u >> 16) & 1u);   // round-to-nearest-even
    return (unsigned short)(r >> 16);
}

// ---------------------------------------------------------------------------
// Kernel 1: fp32 -> bf16 bulk convert (for x)
// ---------------------------------------------------------------------------
__global__ void f32_to_bf16_kernel(const float* __restrict__ in,
                                   unsigned short* __restrict__ out, int n) {
    int i = blockIdx.x * blockDim.x + threadIdx.x;
    if (i < n) out[i] = f32_to_bf16_rne(in[i]);
}

// ---------------------------------------------------------------------------
// Kernel 2: C_bf16[S,E] = A_bf16[S,E] @ W_f32[E,E] + bias_f32[E]
// Block = 128 threads (4 waves). Block tile 64(M) x 64(N); each wave owns a
// 16x64 strip. W k-slab (32x64) is converted to bf16 and staged TRANSPOSED in
// LDS so B-fragments are contiguous 16B runs. Four named B fragments are
// materialized (pure b128 loads + bit_cast, no element moves), then the four
// v_wmma issue back-to-back.
// ---------------------------------------------------------------------------
__global__ __launch_bounds__(128)
void qkv_gemm_bf16(const unsigned short* __restrict__ A,
                   const float* __restrict__ W,
                   const float* __restrict__ bias,
                   unsigned short* __restrict__ Cout) {
    __shared__ unsigned short WtT[64][40];  // [n_local][k_local], 80B row stride (16B aligned)

    const int tid  = threadIdx.x;
    const int lane = tid & 31;
    const int wave = tid >> 5;
    const int half = lane >> 4;
    const int ln   = lane & 15;
    const int n0   = blockIdx.x * 64;
    const int m0   = blockIdx.y * 64 + wave * 16;

    v8f acc[4];
    for (int t = 0; t < 4; ++t)
        for (int r = 0; r < 8; ++r) acc[t][r] = 0.0f;

    const int krow = tid >> 2;          // 0..31 : k-row this thread stages
    const int ncol = (tid & 3) * 16;    // 0,16,32,48 : n-column base (16 floats)

    for (int k0 = 0; k0 < E_DIM; k0 += 32) {
        // --- cooperative stage: W[k0+krow][n0+ncol .. +15] -> WtT[n][k] (bf16)
        const float* wsrc = W + (size_t)(k0 + krow) * E_DIM + n0 + ncol;
        {
            const float4* w4 = (const float4*)wsrc;
            for (int j = 0; j < 4; ++j) {
                float4 q = w4[j];
                WtT[ncol + j * 4 + 0][krow] = f32_to_bf16_rne(q.x);
                WtT[ncol + j * 4 + 1][krow] = f32_to_bf16_rne(q.y);
                WtT[ncol + j * 4 + 2][krow] = f32_to_bf16_rne(q.z);
                WtT[ncol + j * 4 + 3][krow] = f32_to_bf16_rne(q.w);
            }
        }
        if (k0 + 32 < E_DIM)  // hide next slab's HBM latency behind WMMA
            __builtin_prefetch(W + (size_t)(k0 + 32 + krow) * E_DIM + n0 + ncol, 0, 1);

        __syncthreads();

        // --- A fragment: rows m0+ln, K run {half*16..+15}
        const unsigned short* arow = A + (size_t)(m0 + ln) * E_DIM + k0;
        v16bf af = make_frag(*(const uint4*)(arow + half * 16),
                             *(const uint4*)(arow + half * 16 + 8));

        // --- four named B fragments (contiguous b128 from transposed tile)
        const unsigned short* bp0 = &WtT[0 * 16 + ln][half * 16];
        const unsigned short* bp1 = &WtT[1 * 16 + ln][half * 16];
        const unsigned short* bp2 = &WtT[2 * 16 + ln][half * 16];
        const unsigned short* bp3 = &WtT[3 * 16 + ln][half * 16];
        v16bf bf0 = make_frag(*(const uint4*)bp0, *(const uint4*)(bp0 + 8));
        v16bf bf1 = make_frag(*(const uint4*)bp1, *(const uint4*)(bp1 + 8));
        v16bf bf2 = make_frag(*(const uint4*)bp2, *(const uint4*)(bp2 + 8));
        v16bf bf3 = make_frag(*(const uint4*)bp3, *(const uint4*)(bp3 + 8));

        // --- WMMA burst
        acc[0] = __builtin_amdgcn_wmma_f32_16x16x32_bf16(
            false, af, false, bf0, (short)0, acc[0], false, false);
        acc[1] = __builtin_amdgcn_wmma_f32_16x16x32_bf16(
            false, af, false, bf1, (short)0, acc[1], false, false);
        acc[2] = __builtin_amdgcn_wmma_f32_16x16x32_bf16(
            false, af, false, bf2, (short)0, acc[2], false, false);
        acc[3] = __builtin_amdgcn_wmma_f32_16x16x32_bf16(
            false, af, false, bf3, (short)0, acc[3], false, false);

        __syncthreads();
    }

    // --- bias + bf16 store (C layout: M = half*8+r, N = lane%16)
    for (int t = 0; t < 4; ++t) {
        float bt = bias[n0 + t * 16 + ln];
        for (int r = 0; r < 8; ++r) {
            float val = acc[t][r] + bt;
            Cout[(size_t)(m0 + half * 8 + r) * E_DIM + n0 + t * 16 + ln] =
                f32_to_bf16_rne(val);
        }
    }
}

// ---------------------------------------------------------------------------
// Kernel 3: fused flash-style attention. One wave per (head, 16-query tile).
// V tiles are staged memory->LDS with global_load_async_to_lds_b128 (ASYNCcnt),
// bypassing VGPRs. Out-of-segment scores are set to 0.0f and participate in
// the (online) softmax over all 2048 keys (faithful to reference).
// ---------------------------------------------------------------------------
__global__ __launch_bounds__(128)
void attention_bf16(const unsigned short* __restrict__ Q,
                    const unsigned short* __restrict__ K,
                    const unsigned short* __restrict__ V,
                    const int* __restrict__ seg,
                    float* __restrict__ O) {
    __shared__ unsigned short Pt[4][16][32];   // per-wave P tile (queries x 32 keys)
    __shared__ unsigned short Vt[4][32][72];   // per-wave V tile, padded row = 144B

    const int tid  = threadIdx.x;
    const int wave = tid >> 5;
    const int lane = tid & 31;
    const int half = lane >> 4;
    const int ln   = lane & 15;

    const int gw    = blockIdx.x * 4 + wave;   // 0..2047
    const int h     = gw & (H_DIM - 1);
    const int qt    = gw >> 4;
    const int qbase = qt * 16;

    // Q A-fragments for K-dim = 64 (two 16x32 slabs), loaded once.
    const unsigned short* qrow = Q + (size_t)(qbase + ln) * E_DIM + h * D_DIM;
    v16bf qf0 = make_frag(*(const uint4*)(qrow + half * 16),
                          *(const uint4*)(qrow + half * 16 + 8));
    v16bf qf1 = make_frag(*(const uint4*)(qrow + 32 + half * 16),
                          *(const uint4*)(qrow + 32 + half * 16 + 8));

    int segq[8];
    for (int r = 0; r < 8; ++r) segq[r] = seg[qbase + half * 8 + r];

    float mrow[8], lrow[8];
    v8f o[4];
    for (int r = 0; r < 8; ++r) { mrow[r] = -1e30f; lrow[r] = 0.0f; }
    for (int t = 0; t < 4; ++t)
        for (int r = 0; r < 8; ++r) o[t][r] = 0.0f;

    for (int j0 = 0; j0 < S_DIM; j0 += 32) {
        // Previous iteration's LDS reads must retire before async writes land
        // on Vt (ASYNCcnt and DScnt are unordered with each other).
        asm volatile("s_wait_dscnt 0" ::: "memory");

        // --- async stage V[j0..j0+31][h*64..+63] -> LDS, one row per lane.
        //     VDST = per-lane wave-relative LDS address (low 32 bits of the
        //     generic pointer per the ISA LDS-aperture mapping).
        {
            const unsigned short* vrow = V + (size_t)(j0 + lane) * E_DIM + h * D_DIM;
            for (int j = 0; j < 8; ++j) {
                unsigned lds_off = (unsigned)(uintptr_t)(&Vt[wave][lane][j * 8]);
                unsigned long long ga = (unsigned long long)(uintptr_t)(vrow + j * 8);
                asm volatile("global_load_async_to_lds_b128 %0, %1, off"
                             :: "v"(lds_off), "v"(ga) : "memory");
            }
        }
        if (j0 + 32 < S_DIM) {  // prefetch next tiles into GL2
            __builtin_prefetch(K + (size_t)(j0 + 32 + lane) * E_DIM + h * D_DIM, 0, 1);
            __builtin_prefetch(V + (size_t)(j0 + 32 + lane) * E_DIM + h * D_DIM, 0, 1);
        }

        // --- scores: two 16x16 tiles (keys j0..+15 and j0+16..+31), K-dim 64
        const unsigned short* kr0 = K + (size_t)(j0 + ln) * E_DIM + h * D_DIM;
        const unsigned short* kr1 = K + (size_t)(j0 + 16 + ln) * E_DIM + h * D_DIM;

        v8f c0, c1;
        for (int r = 0; r < 8; ++r) { c0[r] = 0.0f; c1[r] = 0.0f; }
        {
            v16bf kb = make_frag(*(const uint4*)(kr0 + half * 16),
                                 *(const uint4*)(kr0 + half * 16 + 8));
            c0 = __builtin_amdgcn_wmma_f32_16x16x32_bf16(
                false, qf0, false, kb, (short)0, c0, false, false);
        }
        {
            v16bf kb = make_frag(*(const uint4*)(kr1 + half * 16),
                                 *(const uint4*)(kr1 + half * 16 + 8));
            c1 = __builtin_amdgcn_wmma_f32_16x16x32_bf16(
                false, qf0, false, kb, (short)0, c1, false, false);
        }
        {
            v16bf kb = make_frag(*(const uint4*)(kr0 + 32 + half * 16),
                                 *(const uint4*)(kr0 + 32 + half * 16 + 8));
            c0 = __builtin_amdgcn_wmma_f32_16x16x32_bf16(
                false, qf1, false, kb, (short)0, c0, false, false);
        }
        {
            v16bf kb = make_frag(*(const uint4*)(kr1 + 32 + half * 16),
                                 *(const uint4*)(kr1 + 32 + half * 16 + 8));
            c1 = __builtin_amdgcn_wmma_f32_16x16x32_bf16(
                false, qf1, false, kb, (short)0, c1, false, false);
        }

        const int sk0 = seg[j0 + ln];
        const int sk1 = seg[j0 + 16 + ln];

        // --- masked online softmax (masked scores = 0.0, still included)
        for (int r = 0; r < 8; ++r) {
            float s0 = (segq[r] == sk0) ? c0[r] : 0.0f;
            float s1 = (segq[r] == sk1) ? c1[r] : 0.0f;
            float tm = fmaxf(s0, s1);
            tm = fmaxf(tm, __shfl_xor(tm, 1));
            tm = fmaxf(tm, __shfl_xor(tm, 2));
            tm = fmaxf(tm, __shfl_xor(tm, 4));
            tm = fmaxf(tm, __shfl_xor(tm, 8));
            float nm = fmaxf(mrow[r], tm);
            float e0 = __expf(s0 - nm);
            float e1 = __expf(s1 - nm);
            float rs = e0 + e1;
            rs += __shfl_xor(rs, 1);
            rs += __shfl_xor(rs, 2);
            rs += __shfl_xor(rs, 4);
            rs += __shfl_xor(rs, 8);
            float sc = __expf(mrow[r] - nm);
            lrow[r]  = lrow[r] * sc + rs;
            mrow[r]  = nm;
            for (int t = 0; t < 4; ++t) o[t][r] *= sc;
            // spill P to LDS (C-layout -> memory row-major)
            Pt[wave][half * 8 + r][ln]      = f32_to_bf16_rne(e0);
            Pt[wave][half * 8 + r][16 + ln] = f32_to_bf16_rne(e1);
        }
        // P stores (DScnt) + async V tile (ASYNCcnt) must be visible.
        asm volatile("s_wait_dscnt 0\n\ts_wait_asynccnt 0" ::: "memory");

        // --- P A-fragment (16x32) from LDS, contiguous 16B runs
        v16bf pf = make_frag(*(const uint4*)&Pt[wave][ln][half * 16],
                             *(const uint4*)&Pt[wave][ln][half * 16 + 8]);

        // --- V B-fragments: strided LDS gather packed into dwords, each
        //     fragment consumed immediately (keeps register pressure low)
        for (int t = 0; t < 4; ++t) {
            v8ui raw;
            for (int i = 0; i < 8; ++i) {
                unsigned lo = Vt[wave][half * 16 + 2 * i + 0][t * 16 + ln];
                unsigned hi = Vt[wave][half * 16 + 2 * i + 1][t * 16 + ln];
                raw[i] = lo | (hi << 16);
            }
            v16bf vbf = __builtin_bit_cast(v16bf, raw);
            o[t] = __builtin_amdgcn_wmma_f32_16x16x32_bf16(
                false, pf, false, vbf, (short)0, o[t], false, false);
        }
    }

    // --- normalize and write attention output (fp32)
    for (int t = 0; t < 4; ++t)
        for (int r = 0; r < 8; ++r)
            O[(size_t)(qbase + half * 8 + r) * E_DIM + h * D_DIM + t * 16 + ln] =
                o[t][r] / lrow[r];
}

// ---------------------------------------------------------------------------
// Kernel 4: u[j] = mean over rows i with seg[i]==seg[pos] of attn[i][j]
// ---------------------------------------------------------------------------
__global__ void seg_mean_kernel(const float* __restrict__ attn,
                                const int* __restrict__ seg,
                                const int* __restrict__ pos,
                                float* __restrict__ u) {
    int j = blockIdx.x * blockDim.x + threadIdx.x;  // 0..E-1
    int sp = seg[pos[0]];
    float sum = 0.0f, cnt = 0.0f;
    for (int i = 0; i < S_DIM; ++i) {
        if (seg[i] == sp) { sum += attn[(size_t)i * E_DIM + j]; cnt += 1.0f; }
    }
    u[j] = sum / cnt;
}

// ---------------------------------------------------------------------------
// Kernel 5: out[j] = bo[j] + sum_e u[e] * Wo[e][j]   (linear readout)
// ---------------------------------------------------------------------------
__global__ void out_gemv_kernel(const float* __restrict__ u,
                                const float* __restrict__ Wo,
                                const float* __restrict__ bo,
                                float* __restrict__ out) {
    int j = blockIdx.x * blockDim.x + threadIdx.x;  // 0..E-1
    float acc = bo[j];
    for (int e = 0; e < E_DIM; ++e)
        acc += u[e] * Wo[(size_t)e * E_DIM + j];
    out[j] = acc;
}

// ---------------------------------------------------------------------------
extern "C" void kernel_launch(void* const* d_in, const int* in_sizes, int n_in,
                              void* d_out, int out_size, void* d_ws, size_t ws_size,
                              hipStream_t stream) {
    const float* x  = (const float*)d_in[0];
    const int*  seg = (const int*)d_in[1];
    const int*  pos = (const int*)d_in[2];
    const float* Wq = (const float*)d_in[3];
    const float* bq = (const float*)d_in[4];
    const float* Wk = (const float*)d_in[5];
    const float* bk = (const float*)d_in[6];
    const float* Wv = (const float*)d_in[7];
    const float* bv = (const float*)d_in[8];
    const float* Wo = (const float*)d_in[9];
    const float* bo = (const float*)d_in[10];
    float* out = (float*)d_out;

    // workspace layout (~25.2 MB)
    char* ws = (char*)d_ws;
    const size_t SE_BF16 = (size_t)S_DIM * E_DIM * sizeof(unsigned short); // 4 MB
    const size_t SE_F32  = (size_t)S_DIM * E_DIM * sizeof(float);          // 8 MB
    unsigned short* xb   = (unsigned short*)(ws);
    unsigned short* Qb   = (unsigned short*)(ws + 1 * SE_BF16);
    unsigned short* Kb   = (unsigned short*)(ws + 2 * SE_BF16);
    unsigned short* Vb   = (unsigned short*)(ws + 3 * SE_BF16);
    float*          attn = (float*)(ws + 4 * SE_BF16);
    float*          u    = (float*)(ws + 4 * SE_BF16 + SE_F32);

    // 1) x -> bf16
    {
        int n = S_DIM * E_DIM;
        f32_to_bf16_kernel<<<(n + 255) / 256, 256, 0, stream>>>(x, xb, n);
    }
    // 2) Q/K/V projections with WMMA bf16 (W converted to bf16 in-kernel)
    {
        dim3 grid(E_DIM / 64, S_DIM / 64);  // (16, 32)
        qkv_gemm_bf16<<<grid, 128, 0, stream>>>(xb, Wq, bq, Qb);
        qkv_gemm_bf16<<<grid, 128, 0, stream>>>(xb, Wk, bk, Kb);
        qkv_gemm_bf16<<<grid, 128, 0, stream>>>(xb, Wv, bv, Vb);
    }
    // 3) fused attention: 2048 (head, qtile) waves, 4 waves/block
    attention_bf16<<<(H_DIM * (S_DIM / 16)) / 4, 128, 0, stream>>>(
        Qb, Kb, Vb, seg, attn);
    // 4) segment mean over attention output rows
    seg_mean_kernel<<<E_DIM / 256, 256, 0, stream>>>(attn, seg, pos, u);
    // 5) linear readout GEMV
    out_gemv_kernel<<<E_DIM / 256, 256, 0, stream>>>(u, Wo, bo, out);
}
